// DGSRLayer_49048526520673
// MI455X (gfx1250) — matrix operand order repaired and checked by reading, not compiled, not run on hardware
//
#include <hip/hip_runtime.h>
#include <math.h>

#define D_DIM 128

typedef __attribute__((ext_vector_type(16))) __bf16 v16bf;
typedef __attribute__((ext_vector_type(8)))  float  v8f;

// ---------------------------------------------------------------------------
// Float atomic max via ordered-integer trick (init value must be -inf).
// ---------------------------------------------------------------------------
__device__ __forceinline__ void atomicMaxF32(float* addr, float val) {
  if (val >= 0.0f) {
    atomicMax((int*)addr, __float_as_int(val));
  } else {
    atomicMin((unsigned int*)addr, (unsigned int)__float_as_int(val));
  }
}

// ---------------------------------------------------------------------------
// Fills
// ---------------------------------------------------------------------------
__global__ void fill_kernel(float* __restrict__ p, long n, float v) {
  long i = (long)blockIdx.x * 256 + threadIdx.x;
  if (i < n) p[i] = v;
}
__global__ void fill_int_kernel(int* __restrict__ p, long n, int v) {
  long i = (long)blockIdx.x * 256 + threadIdx.x;
  if (i < n) p[i] = v;
}
__global__ void set_int_kernel(int* __restrict__ p, int v) { *p = v; }

// ---------------------------------------------------------------------------
// f32 -> bf16 weights (round to nearest even)
// ---------------------------------------------------------------------------
__global__ void cvt_bf16_kernel(const float* __restrict__ src,
                                unsigned short* __restrict__ dst, int n) {
  int i = blockIdx.x * 256 + threadIdx.x;
  if (i < n) {
    unsigned int u = __float_as_uint(src[i]);
    unsigned int r = (u + 0x7fffu + ((u >> 16) & 1u)) >> 16;
    dst[i] = (unsigned short)r;
  }
}

// ---------------------------------------------------------------------------
// WMMA GEMM:  Y[M,128] = X'[M,128] @ W^T   (W is 128x128 bf16 row-major [n][k])
// X' row m = X[gatherIdx ? gatherIdx[m] : m].  One wave per 16-row tile.
// ---------------------------------------------------------------------------
__global__ __launch_bounds__(256)
void gemm_bf16_wmma_kernel(const float* __restrict__ X,
                           const unsigned short* __restrict__ Wb,
                           const int* __restrict__ gatherIdx,
                           float* __restrict__ Y, int M) {
  __shared__ alignas(32) __bf16 ldsW[128 * 128];
  {
    const uint4* src = (const uint4*)Wb;
    uint4* dst = (uint4*)ldsW;
    for (int t = threadIdx.x; t < 2048; t += 256) dst[t] = src[t];
  }
  __syncthreads();

  const int lane = threadIdx.x & 31;
  const int wave = threadIdx.x >> 5;
  const int rowBase = (blockIdx.x * 8 + wave) * 16;
  if (rowBase >= M) return;

  const int grp = lane >> 4;
  const int m = rowBase + (lane & 15);
  const long srcRow = gatherIdx ? (long)gatherIdx[m] : (long)m;
  const float* xrow = X + srcRow * D_DIM;

  v8f acc[8];
  const v8f zero = {0.f, 0.f, 0.f, 0.f, 0.f, 0.f, 0.f, 0.f};
#pragma unroll
  for (int n = 0; n < 8; ++n) acc[n] = zero;

#pragma unroll
  for (int k0 = 0; k0 < 128; k0 += 32) {
    v16bf a;
    const int off = k0 + grp * 16;
#pragma unroll
    for (int j = 0; j < 8; ++j) {
      a[j]     = (__bf16)xrow[off + j];
      a[j + 8] = (__bf16)xrow[off + 8 + j];
    }
    const int koff = k0 + grp * 16;
#pragma unroll
    for (int n = 0; n < 8; ++n) {
      const int col = n * 16 + (lane & 15);
      const v16bf b = *(const v16bf*)&ldsW[col * 128 + koff];
      acc[n] = __builtin_amdgcn_wmma_f32_16x16x32_bf16(
          false, a, false, b, (short)0, acc[n], false, false);
    }
  }

  const int nOff = lane & 15;
  const int rOff = grp * 8;
#pragma unroll
  for (int n = 0; n < 8; ++n) {
#pragma unroll
    for (int j = 0; j < 8; ++j) {
      Y[(long)(rowBase + rOff + j) * D_DIM + n * 16 + nOff] = acc[n][j];
    }
  }
}

// ---------------------------------------------------------------------------
// CSR construction: histogram, per-block exclusive scan, partials scan,
// offset add, bucket fill.
// ---------------------------------------------------------------------------
__global__ void count_kernel(const int* __restrict__ seg, int* __restrict__ counts,
                             int E) {
  int e = blockIdx.x * 256 + threadIdx.x;
  if (e < E) atomicAdd(&counts[seg[e]], 1);
}

__global__ __launch_bounds__(256)
void scan_block_kernel(const int* __restrict__ counts, int* __restrict__ offs,
                       int* __restrict__ partial, int n) {
  __shared__ int lds[256];
  const int i = blockIdx.x * 256 + threadIdx.x;
  const int v = (i < n) ? counts[i] : 0;
  lds[threadIdx.x] = v;
  __syncthreads();
  for (int o = 1; o < 256; o <<= 1) {
    int t = (threadIdx.x >= (unsigned)o) ? lds[threadIdx.x - o] : 0;
    __syncthreads();
    lds[threadIdx.x] += t;
    __syncthreads();
  }
  if (i < n) offs[i] = lds[threadIdx.x] - v;   // exclusive
  if (threadIdx.x == 255) partial[blockIdx.x] = lds[255];
}

__global__ __launch_bounds__(256)
void scan_partials_kernel(int* __restrict__ partial, int nb) {
  __shared__ int lds[256];
  __shared__ int carry;
  if (threadIdx.x == 0) carry = 0;
  __syncthreads();
  for (int base = 0; base < nb; base += 256) {
    const int i = base + threadIdx.x;
    const int v = (i < nb) ? partial[i] : 0;
    lds[threadIdx.x] = v;
    __syncthreads();
    for (int o = 1; o < 256; o <<= 1) {
      int t = (threadIdx.x >= (unsigned)o) ? lds[threadIdx.x - o] : 0;
      __syncthreads();
      lds[threadIdx.x] += t;
      __syncthreads();
    }
    const int inc = lds[threadIdx.x];
    if (i < nb) partial[i] = carry + inc - v;  // exclusive across blocks
    __syncthreads();
    if (threadIdx.x == 255) carry += lds[255];
    __syncthreads();
  }
}

__global__ void scan_add_kernel(int* __restrict__ offs,
                                const int* __restrict__ partial, int n) {
  int i = blockIdx.x * 256 + threadIdx.x;
  if (i < n) offs[i] += partial[blockIdx.x];
}

__global__ void bucket_fill_kernel(const int* __restrict__ seg,
                                   int* __restrict__ cursor,
                                   int* __restrict__ list, int E) {
  int e = blockIdx.x * 256 + threadIdx.x;
  if (e < E) {
    int pos = atomicAdd(&cursor[seg[e]], 1);
    list[pos] = e;
  }
}

// ---------------------------------------------------------------------------
// Edge scores: 5 dot products per edge (wave per edge) + segmented atomic max.
// ---------------------------------------------------------------------------
__global__ __launch_bounds__(256)
void edge_scores_kernel(const int* __restrict__ eu, const int* __restrict__ ei,
                        const int* __restrict__ rui, const int* __restrict__ riu,
                        const float* __restrict__ um, const float* __restrict__ im,
                        const float* __restrict__ li, const float* __restrict__ lu,
                        const float* __restrict__ pV, const float* __restrict__ pK,
                        float* __restrict__ sUI, float* __restrict__ sIU,
                        float* __restrict__ sA, float* __restrict__ sB,
                        float* __restrict__ maxUL, float* __restrict__ maxIL,
                        float* __restrict__ maxUS, float* __restrict__ maxIS,
                        int E) {
  const int lane = threadIdx.x & 31;
  const int e = blockIdx.x * 8 + (threadIdx.x >> 5);
  if (e >= E) return;
  const int u = eu[e], it = ei[e];
  const float4 a  = ((const float4*)(um + (long)u  * D_DIM))[lane];
  const float4 b  = ((const float4*)(im + (long)it * D_DIM))[lane];
  const float4 pv = ((const float4*)(pV + (long)rui[e] * D_DIM))[lane];
  const float4 pk = ((const float4*)(pK + (long)riu[e] * D_DIM))[lane];
  const float4 l1 = ((const float4*)(li + (long)u * D_DIM))[lane];
  const float4 l2 = ((const float4*)(lu + (long)u * D_DIM))[lane];

  float d0 = a.x*b.x + a.y*b.y + a.z*b.z + a.w*b.w;
  float d1 = a.x*pv.x + a.y*pv.y + a.z*pv.z + a.w*pv.w;
  float d2 = b.x*pk.x + b.y*pk.y + b.z*pk.z + b.w*pk.w;
  float d3 = l1.x*b.x + l1.y*b.y + l1.z*b.z + l1.w*b.w;
  float d4 = l2.x*b.x + l2.y*b.y + l2.z*b.z + l2.w*b.w;
#pragma unroll
  for (int o = 16; o > 0; o >>= 1) {
    d0 += __shfl_xor(d0, o, 32);
    d1 += __shfl_xor(d1, o, 32);
    d2 += __shfl_xor(d2, o, 32);
    d3 += __shfl_xor(d3, o, 32);
    d4 += __shfl_xor(d4, o, 32);
  }
  if (lane == 0) {
    const float s = 0.08838834764831845f;  // 1/sqrt(128)
    const float eui = (d0 + d1) * s;
    const float eiu = (d0 + d2) * s;
    const float av = d3 * s;
    const float bv = d4 * s;
    sUI[e] = eui; sIU[e] = eiu; sA[e] = av; sB[e] = bv;
    atomicMaxF32(&maxUL[u], eui);
    atomicMaxF32(&maxIL[it], eiu);
    atomicMaxF32(&maxUS[u], av);
    atomicMaxF32(&maxIS[it], bv);
  }
}

// ---------------------------------------------------------------------------
// exp(score - segmax) in place + atomic segmented sums (scalar, cheap)
// ---------------------------------------------------------------------------
__global__ __launch_bounds__(256)
void edge_expsum_kernel(const int* __restrict__ eu, const int* __restrict__ ei,
                        float* __restrict__ sUI, float* __restrict__ sIU,
                        float* __restrict__ sA, float* __restrict__ sB,
                        const float* __restrict__ maxUL, const float* __restrict__ maxIL,
                        const float* __restrict__ maxUS, const float* __restrict__ maxIS,
                        float* __restrict__ sumUL, float* __restrict__ sumIL,
                        float* __restrict__ sumUS, float* __restrict__ sumIS,
                        int E) {
  const int e = blockIdx.x * 256 + threadIdx.x;
  if (e >= E) return;
  const int u = eu[e], it = ei[e];
  float x;
  x = __expf(sUI[e] - maxUL[u]);  sUI[e] = x; atomicAdd(&sumUL[u], x);
  x = __expf(sIU[e] - maxIL[it]); sIU[e] = x; atomicAdd(&sumIL[it], x);
  x = __expf(sA[e]  - maxUS[u]);  sA[e]  = x; atomicAdd(&sumUS[u], x);
  x = __expf(sB[e]  - maxIS[it]); sB[e]  = x; atomicAdd(&sumIS[it], x);
}

// ---------------------------------------------------------------------------
// Gather-style message passing: one wave per user; register accumulation;
// single vector store per output row (no float atomics).
// hLu[u] = sum_e aL*(imb[i]+pK[riu]);  hSu[u] = sum_e aS*(im[i]+1)
// ---------------------------------------------------------------------------
__global__ __launch_bounds__(256)
void gather_user_kernel(const int* __restrict__ offs, const int* __restrict__ list,
                        const int* __restrict__ ei, const int* __restrict__ riu,
                        const float* __restrict__ imb, const float* __restrict__ im,
                        const float* __restrict__ pK,
                        const float* __restrict__ sUI, const float* __restrict__ sA,
                        const float* __restrict__ sumUL, const float* __restrict__ sumUS,
                        float* __restrict__ hLu, float* __restrict__ hSu, int NU) {
  const int lane = threadIdx.x & 31;
  const int u = blockIdx.x * 8 + (threadIdx.x >> 5);
  if (u >= NU) return;
  const int start = offs[u], end = offs[u + 1];
  float4 accL = {0.f, 0.f, 0.f, 0.f};
  float4 accS = {0.f, 0.f, 0.f, 0.f};
  if (end > start) {
    const float invL = 1.0f / sumUL[u];
    const float invS = 1.0f / sumUS[u];
    for (int p = start; p < end; ++p) {
      const int e = list[p];
      const float aL = sUI[e] * invL;
      const float aS = sA[e] * invS;
      const int i = ei[e];
      const int r = riu[e];
      const float4 mb = ((const float4*)(imb + (long)i * D_DIM))[lane];
      const float4 pk = ((const float4*)(pK + (long)r * D_DIM))[lane];
      const float4 mi = ((const float4*)(im + (long)i * D_DIM))[lane];
      accL.x += aL * (mb.x + pk.x); accL.y += aL * (mb.y + pk.y);
      accL.z += aL * (mb.z + pk.z); accL.w += aL * (mb.w + pk.w);
      accS.x += aS * (mi.x + 1.0f); accS.y += aS * (mi.y + 1.0f);
      accS.z += aS * (mi.z + 1.0f); accS.w += aS * (mi.w + 1.0f);
    }
  }
  ((float4*)(hLu + (long)u * D_DIM))[lane] = accL;
  ((float4*)(hSu + (long)u * D_DIM))[lane] = accS;
}

// hLi[i] = sum_e bL*(umb[u]+pV[rui]);  hSi[i] = sum_e bS*(um[u]+1)
__global__ __launch_bounds__(256)
void gather_item_kernel(const int* __restrict__ offs, const int* __restrict__ list,
                        const int* __restrict__ eu, const int* __restrict__ rui,
                        const float* __restrict__ umb, const float* __restrict__ um,
                        const float* __restrict__ pV,
                        const float* __restrict__ sIU, const float* __restrict__ sB,
                        const float* __restrict__ sumIL, const float* __restrict__ sumIS,
                        float* __restrict__ hLi, float* __restrict__ hSi, int NI) {
  const int lane = threadIdx.x & 31;
  const int it = blockIdx.x * 8 + (threadIdx.x >> 5);
  if (it >= NI) return;
  const int start = offs[it], end = offs[it + 1];
  float4 accL = {0.f, 0.f, 0.f, 0.f};
  float4 accS = {0.f, 0.f, 0.f, 0.f};
  if (end > start) {
    const float invL = 1.0f / sumIL[it];
    const float invS = 1.0f / sumIS[it];
    for (int p = start; p < end; ++p) {
      const int e = list[p];
      const float bL = sIU[e] * invL;
      const float bS = sB[e] * invS;
      const int u = eu[e];
      const int r = rui[e];
      const float4 mb = ((const float4*)(umb + (long)u * D_DIM))[lane];
      const float4 pv = ((const float4*)(pV + (long)r * D_DIM))[lane];
      const float4 mu = ((const float4*)(um + (long)u * D_DIM))[lane];
      accL.x += bL * (mb.x + pv.x); accL.y += bL * (mb.y + pv.y);
      accL.z += bL * (mb.z + pv.z); accL.w += bL * (mb.w + pv.w);
      accS.x += bS * (mu.x + 1.0f); accS.y += bS * (mu.y + 1.0f);
      accS.z += bS * (mu.z + 1.0f); accS.w += bS * (mu.w + 1.0f);
    }
  }
  ((float4*)(hLi + (long)it * D_DIM))[lane] = accL;
  ((float4*)(hSi + (long)it * D_DIM))[lane] = accS;
}

// ---------------------------------------------------------------------------
// Host launcher
// ---------------------------------------------------------------------------
extern "C" void kernel_launch(void* const* d_in, const int* in_sizes, int n_in,
                              void* d_out, int out_size, void* d_ws, size_t ws_size,
                              hipStream_t stream) {
  const float* u_emb = (const float*)d_in[0];
  const float* i_emb = (const float*)d_in[1];
  const int* edge_u = (const int*)d_in[2];
  const int* edge_i = (const int*)d_in[3];
  const int* rui = (const int*)d_in[4];
  const int* riu = (const int*)d_in[5];
  const int* last_u_items = (const int*)d_in[6];
  const int* last_i_users = (const int*)d_in[7];
  const float* W1  = (const float*)d_in[8];
  const float* W2  = (const float*)d_in[9];
  const float* W1b = (const float*)d_in[10];
  const float* W2b = (const float*)d_in[11];
  const float* W3  = (const float*)d_in[12];
  const float* W4  = (const float*)d_in[13];
  const float* pV  = (const float*)d_in[14];
  const float* pK  = (const float*)d_in[15];
  const float* last_user_table = (const float*)d_in[16];
  const float* last_item_table = (const float*)d_in[17];

  const long NU = in_sizes[0] / D_DIM;
  const long NI = in_sizes[1] / D_DIM;
  const int  E  = in_sizes[2];

  // ---- workspace carve-out (256B aligned) ----
  char* ws = (char*)d_ws;
  size_t off = 0;
  auto alloc = [&](size_t bytes) -> void* {
    void* p = ws + off;
    off = (off + bytes + 255) & ~(size_t)255;
    return p;
  };
  unsigned short* wb = (unsigned short*)alloc((size_t)6 * 128 * 128 * 2);
  float* um  = (float*)alloc((size_t)NU * D_DIM * 4);
  float* im  = (float*)alloc((size_t)NI * D_DIM * 4);
  float* umb = (float*)alloc((size_t)NU * D_DIM * 4);
  float* imb = (float*)alloc((size_t)NI * D_DIM * 4);
  float* li  = (float*)alloc((size_t)NU * D_DIM * 4);
  float* lu  = (float*)alloc((size_t)NU * D_DIM * 4);
  float* sUI = (float*)alloc((size_t)E * 4);
  float* sIU = (float*)alloc((size_t)E * 4);
  float* sA  = (float*)alloc((size_t)E * 4);
  float* sB  = (float*)alloc((size_t)E * 4);
  float* maxBase = (float*)alloc((size_t)(2 * NU + 2 * NI) * 4);
  float* maxUL = maxBase;
  float* maxIL = maxUL + NU;
  float* maxUS = maxIL + NI;
  float* maxIS = maxUS + NU;
  float* sumBase = (float*)alloc((size_t)(2 * NU + 2 * NI) * 4);
  float* sumUL = sumBase;
  float* sumIL = sumUL + NU;
  float* sumUS = sumIL + NI;
  float* sumIS = sumUS + NU;
  int* cntU  = (int*)alloc((size_t)NU * 4);        // also reused as cursor
  int* cntI  = (int*)alloc((size_t)NI * 4);
  int* offsU = (int*)alloc((size_t)(NU + 1) * 4);
  int* offsI = (int*)alloc((size_t)(NI + 1) * 4);
  int* listU = (int*)alloc((size_t)E * 4);
  int* listI = (int*)alloc((size_t)E * 4);
  const int nbU = (int)((NU + 255) / 256);
  const int nbI = (int)((NI + 255) / 256);
  int* partial = (int*)alloc((size_t)(nbU > nbI ? nbU : nbI) * 4);

  float* out = (float*)d_out;
  float* hLu = out;
  float* hSu = out + NU * D_DIM;
  float* hLi = out + 2 * NU * D_DIM;
  float* hSi = out + 2 * NU * D_DIM + NI * D_DIM;

  // ---- init (no output fill needed: gather kernels write every row) ----
  {
    const long nm = 2 * NU + 2 * NI;
    fill_kernel<<<(unsigned)((nm + 255) / 256), 256, 0, stream>>>(maxBase, nm, -INFINITY);
    fill_kernel<<<(unsigned)((nm + 255) / 256), 256, 0, stream>>>(sumBase, nm, 0.0f);
    fill_int_kernel<<<(unsigned)((NU + 255) / 256), 256, 0, stream>>>(cntU, NU, 0);
    fill_int_kernel<<<(unsigned)((NI + 255) / 256), 256, 0, stream>>>(cntI, NI, 0);
  }

  // ---- weights -> bf16 ----
  const int WN = 128 * 128;
  cvt_bf16_kernel<<<(WN + 255) / 256, 256, 0, stream>>>(W2,  wb + 0 * WN, WN);
  cvt_bf16_kernel<<<(WN + 255) / 256, 256, 0, stream>>>(W1,  wb + 1 * WN, WN);
  cvt_bf16_kernel<<<(WN + 255) / 256, 256, 0, stream>>>(W2b, wb + 2 * WN, WN);
  cvt_bf16_kernel<<<(WN + 255) / 256, 256, 0, stream>>>(W1b, wb + 3 * WN, WN);
  cvt_bf16_kernel<<<(WN + 255) / 256, 256, 0, stream>>>(W3,  wb + 4 * WN, WN);
  cvt_bf16_kernel<<<(WN + 255) / 256, 256, 0, stream>>>(W4,  wb + 5 * WN, WN);

  // ---- six GEMMs on the matrix cores ----
  const unsigned gU = (unsigned)((NU + 127) / 128);
  const unsigned gI = (unsigned)((NI + 127) / 128);
  gemm_bf16_wmma_kernel<<<gU, 256, 0, stream>>>(u_emb, wb + 0 * WN, nullptr, um,  (int)NU);
  gemm_bf16_wmma_kernel<<<gI, 256, 0, stream>>>(i_emb, wb + 1 * WN, nullptr, im,  (int)NI);
  gemm_bf16_wmma_kernel<<<gU, 256, 0, stream>>>(u_emb, wb + 2 * WN, nullptr, umb, (int)NU);
  gemm_bf16_wmma_kernel<<<gI, 256, 0, stream>>>(i_emb, wb + 3 * WN, nullptr, imb, (int)NI);
  gemm_bf16_wmma_kernel<<<gU, 256, 0, stream>>>(last_item_table, wb + 4 * WN, last_u_items, li, (int)NU);
  gemm_bf16_wmma_kernel<<<gU, 256, 0, stream>>>(last_user_table, wb + 5 * WN, last_i_users, lu, (int)NU);

  // ---- CSR build for both segmentations ----
  const unsigned gE256 = (unsigned)((E + 255) / 256);
  count_kernel<<<gE256, 256, 0, stream>>>(edge_u, cntU, E);
  count_kernel<<<gE256, 256, 0, stream>>>(edge_i, cntI, E);

  scan_block_kernel<<<(unsigned)nbU, 256, 0, stream>>>(cntU, offsU, partial, (int)NU);
  scan_partials_kernel<<<1, 256, 0, stream>>>(partial, nbU);
  scan_add_kernel<<<(unsigned)nbU, 256, 0, stream>>>(offsU, partial, (int)NU);
  set_int_kernel<<<1, 1, 0, stream>>>(offsU + NU, E);

  scan_block_kernel<<<(unsigned)nbI, 256, 0, stream>>>(cntI, offsI, partial, (int)NI);
  scan_partials_kernel<<<1, 256, 0, stream>>>(partial, nbI);
  scan_add_kernel<<<(unsigned)nbI, 256, 0, stream>>>(offsI, partial, (int)NI);
  set_int_kernel<<<1, 1, 0, stream>>>(offsI + NI, E);

  // cursor := offsets (reuse count buffers), then bucket fill
  hipMemcpyAsync(cntU, offsU, (size_t)NU * 4, hipMemcpyDeviceToDevice, stream);
  hipMemcpyAsync(cntI, offsI, (size_t)NI * 4, hipMemcpyDeviceToDevice, stream);
  bucket_fill_kernel<<<gE256, 256, 0, stream>>>(edge_u, cntU, listU, E);
  bucket_fill_kernel<<<gE256, 256, 0, stream>>>(edge_i, cntI, listI, E);

  // ---- edge scores + softmax stats ----
  const unsigned gE8 = (unsigned)((E + 7) / 8);
  edge_scores_kernel<<<gE8, 256, 0, stream>>>(edge_u, edge_i, rui, riu,
                                              um, im, li, lu, pV, pK,
                                              sUI, sIU, sA, sB,
                                              maxUL, maxIL, maxUS, maxIS, E);
  edge_expsum_kernel<<<gE256, 256, 0, stream>>>(edge_u, edge_i, sUI, sIU, sA, sB,
                                                maxUL, maxIL, maxUS, maxIS,
                                                sumUL, sumIL, sumUS, sumIS, E);

  // ---- gather-style message passing (no float atomics) ----
  gather_user_kernel<<<(unsigned)((NU + 7) / 8), 256, 0, stream>>>(
      offsU, listU, edge_i, riu, imb, im, pK,
      sUI, sA, sumUL, sumUS, hLu, hSu, (int)NU);
  gather_item_kernel<<<(unsigned)((NI + 7) / 8), 256, 0, stream>>>(
      offsI, listI, edge_u, rui, umb, um, pV,
      sIU, sB, sumIL, sumIS, hLi, hSi, (int)NI);
}
